// CRF_RNN_Layer_57569741635967
// MI455X (gfx1250) — matrix operand order repaired and checked by reading, not compiled, not run on hardware
//
#include <hip/hip_runtime.h>

// ---------------------------------------------------------------------------
// CRF-RNN mean-field for MI455X (gfx1250, wave32, WMMA).
// Precompute K_spatial / K_bilateral once in bf16 (streaming ~340MB/iter,
// memory-bound -> ~73us/5iters at 23.3TB/s), then 5 iterations of
//   softmax -> [K_s@q, K_b@q via v_wmma_f32_16x16x32_bf16] -> fused epilogue.
// Occupancy: k-dim split 8 ways per row tile (8 waves/block, 576 blocks ->
// 4608 waves), partials combined via ds_add_f32 LDS atomics.
// ---------------------------------------------------------------------------

#define HH 96
#define WW 96
#define CC 21
#define NPIX 9216           // HH*WW
#define CPAD 32             // C padded to 2 WMMA column tiles
#define ITERS 5
#define NW 8                // waves per block in crf_step
#define KCH (NPIX / NW)     // 1152 k-elements per wave

typedef __attribute__((ext_vector_type(8)))  __bf16 v8bf;
typedef __attribute__((ext_vector_type(16))) __bf16 v16bf;
typedef __attribute__((ext_vector_type(8)))  float  v8f;

__device__ __forceinline__ unsigned f2bf_bits(float f) {
    union { float f; unsigned u; } v; v.f = f;
    return (v.u + 0x7FFFu + ((v.u >> 16) & 1u)) >> 16;   // RNE
}

__device__ __forceinline__ v16bf cat16(v8bf lo, v8bf hi) {
    return __builtin_shufflevector(lo, hi, 0,1,2,3,4,5,6,7,8,9,10,11,12,13,14,15);
}

// --------------------------------------------------------------------------
// 1) Per-pixel features (SoA): fs0 fs1 ns fb0..fb4 nb  (9 floats per pixel)
// --------------------------------------------------------------------------
__global__ void feats_kernel(const float* __restrict__ img, float* __restrict__ feat) {
    int n = blockIdx.x * 256 + threadIdx.x;
    if (n >= NPIX) return;
    float y = (float)(n / WW), x = (float)(n % WW);
    float fs0 = y * (1.0f / 3.0f), fs1 = x * (1.0f / 3.0f);   // / theta_gamma
    float fb0 = y * 0.125f, fb1 = x * 0.125f;                 // / theta_alpha
    float r = img[n * 3 + 0] * 8.0f;                          // / theta_beta
    float g = img[n * 3 + 1] * 8.0f;
    float b = img[n * 3 + 2] * 8.0f;
    feat[0 * NPIX + n] = fs0;
    feat[1 * NPIX + n] = fs1;
    feat[2 * NPIX + n] = fs0 * fs0 + fs1 * fs1;
    feat[3 * NPIX + n] = fb0;
    feat[4 * NPIX + n] = fb1;
    feat[5 * NPIX + n] = r;
    feat[6 * NPIX + n] = g;
    feat[7 * NPIX + n] = b;
    feat[8 * NPIX + n] = fb0 * fb0 + fb1 * fb1 + r * r + g * g + b * b;
}

// --------------------------------------------------------------------------
// 2) Build bf16 Gaussian kernels K[i,j] = exp(-0.5*||f_i - f_j||^2).
//    Two j per thread -> packed 32-bit stores.
// --------------------------------------------------------------------------
__global__ void buildk_kernel(const float* __restrict__ feat,
                              unsigned* __restrict__ Ks,
                              unsigned* __restrict__ Kb) {
    int jp = blockIdx.x * 256 + threadIdx.x;      // pair index
    int i  = blockIdx.y;                          // uniform -> scalar loads
    float fs0i = feat[0 * NPIX + i], fs1i = feat[1 * NPIX + i], nsi = feat[2 * NPIX + i];
    float fb0i = feat[3 * NPIX + i], fb1i = feat[4 * NPIX + i];
    float ri = feat[5 * NPIX + i], gi = feat[6 * NPIX + i], bi = feat[7 * NPIX + i];
    float nbi = feat[8 * NPIX + i];

    unsigned ks2 = 0, kb2 = 0;
    #pragma unroll
    for (int h = 0; h < 2; ++h) {
        int j = 2 * jp + h;
        float d2s = nsi + feat[2 * NPIX + j]
                  - 2.0f * (fs0i * feat[0 * NPIX + j] + fs1i * feat[1 * NPIX + j]);
        float d2b = nbi + feat[8 * NPIX + j]
                  - 2.0f * (fb0i * feat[3 * NPIX + j] + fb1i * feat[4 * NPIX + j]
                          + ri * feat[5 * NPIX + j] + gi * feat[6 * NPIX + j]
                          + bi * feat[7 * NPIX + j]);
        float ks = __expf(-0.5f * fmaxf(d2s, 0.0f));
        float kb = __expf(-0.5f * fmaxf(d2b, 0.0f));
        ks2 |= f2bf_bits(ks) << (16 * h);
        kb2 |= f2bf_bits(kb) << (16 * h);
    }
    size_t idx = (size_t)i * (NPIX / 2) + jp;
    Ks[idx] = ks2;
    Kb[idx] = kb2;
}

// --------------------------------------------------------------------------
// 3) Softmax over C; store q transposed bf16: qT[c*NPIX + n], rows 21..31 = 0.
// --------------------------------------------------------------------------
__global__ void softmax_kernel(const float* __restrict__ q, unsigned short* __restrict__ qT) {
    int n = blockIdx.x * 256 + threadIdx.x;
    if (n >= NPIX) return;
    float v[CC];
    float m = -3.4e38f;
    #pragma unroll
    for (int c = 0; c < CC; ++c) { v[c] = q[n * CC + c]; m = fmaxf(m, v[c]); }
    float s = 0.0f;
    #pragma unroll
    for (int c = 0; c < CC; ++c) { v[c] = __expf(v[c] - m); s += v[c]; }
    float inv = 1.0f / s;
    #pragma unroll
    for (int c = 0; c < CC; ++c) qT[c * NPIX + n] = (unsigned short)f2bf_bits(v[c] * inv);
    #pragma unroll
    for (int c = CC; c < CPAD; ++c) qT[c * NPIX + n] = 0;
}

// --------------------------------------------------------------------------
// 4) One mean-field step. One block (8 waves) per 16-row output tile; wave w
//    accumulates k in [w*KCH, (w+1)*KCH) with 4 WMMA accumulators, partials
//    merged with ds_add_f32, epilogue fused.
// --------------------------------------------------------------------------
__global__ void __launch_bounds__(256)
crf_step_kernel(const unsigned short* __restrict__ Ks,
                const unsigned short* __restrict__ Kb,
                const unsigned short* __restrict__ qT,
                const float* __restrict__ u,
                const float* __restrict__ sw,
                const float* __restrict__ bw,
                const float* __restrict__ compat,
                float* __restrict__ qout) {
    __shared__ float accS[16][33];         // +1 pad: stagger LDS banks
    __shared__ float accB[16][33];
    __shared__ float smsg[16][22];
    __shared__ float scompat[CC * CC];

    for (int t = threadIdx.x; t < 16 * 33; t += 256) {
        (&accS[0][0])[t] = 0.0f;
        (&accB[0][0])[t] = 0.0f;
    }
    for (int t = threadIdx.x; t < CC * CC; t += 256) scompat[t] = compat[t];
    __syncthreads();

    const int wave = threadIdx.x >> 5;
    const int lane = threadIdx.x & 31;
    const int i0   = blockIdx.x * 16;                // 576 blocks = 576 tiles
    const int mrow = lane & 15;                      // A row / B column in tile
    const int g    = lane >> 4;                      // lane half
    const int kbeg = wave * KCH;

    // A (16x32 bf16): lane holds row M=mrow, K = 8g+{0..7} and 8g+16+{0..7}
    const unsigned short* As = Ks + (size_t)(i0 + mrow) * NPIX + 8 * g + kbeg;
    const unsigned short* Ab = Kb + (size_t)(i0 + mrow) * NPIX + 8 * g + kbeg;
    // B (32x16 bf16): lane holds col N=mrow, K = 8g+{0..7} and 8g+16+{0..7}
    const unsigned short* B0 = qT + (size_t)mrow * NPIX + 8 * g + kbeg;         // c 0..15
    const unsigned short* B1 = qT + (size_t)(mrow + 16) * NPIX + 8 * g + kbeg;  // c 16..31

    v8f acc_s0 = {}, acc_s1 = {}, acc_b0 = {}, acc_b1 = {};

    #pragma unroll 2
    for (int k0 = 0; k0 < KCH; k0 += 32) {
        v8bf asl = *(const v8bf*)(As + k0);
        v8bf ash = *(const v8bf*)(As + k0 + 16);
        v8bf abl = *(const v8bf*)(Ab + k0);
        v8bf abh = *(const v8bf*)(Ab + k0 + 16);
        v8bf b0l = *(const v8bf*)(B0 + k0);
        v8bf b0h = *(const v8bf*)(B0 + k0 + 16);
        v8bf b1l = *(const v8bf*)(B1 + k0);
        v8bf b1h = *(const v8bf*)(B1 + k0 + 16);
        // stream K ahead (8 k-steps): global_prefetch_b8
        __builtin_prefetch(As + k0 + 256, 0, 1);
        __builtin_prefetch(Ab + k0 + 256, 0, 1);

        v16bf a_s = cat16(asl, ash);
        v16bf a_b = cat16(abl, abh);
        v16bf b_0 = cat16(b0l, b0h);
        v16bf b_1 = cat16(b1l, b1h);

        acc_s0 = __builtin_amdgcn_wmma_f32_16x16x32_bf16(false, a_s, false, b_0, (short)0, acc_s0, false, false);
        acc_s1 = __builtin_amdgcn_wmma_f32_16x16x32_bf16(false, a_s, false, b_1, (short)0, acc_s1, false, false);
        acc_b0 = __builtin_amdgcn_wmma_f32_16x16x32_bf16(false, a_b, false, b_0, (short)0, acc_b0, false, false);
        acc_b1 = __builtin_amdgcn_wmma_f32_16x16x32_bf16(false, a_b, false, b_1, (short)0, acc_b1, false, false);
    }

    // D layout: vgpr r, lane (g, mrow) -> row m = r + 8g, col mrow / mrow+16.
    // Merge the 8 per-wave partials with LDS float atomics (ds_add_f32).
    #pragma unroll
    for (int r = 0; r < 8; ++r) {
        int m = r + 8 * g;
        atomicAdd(&accS[m][mrow],      acc_s0[r]);
        atomicAdd(&accS[m][mrow + 16], acc_s1[r]);
        atomicAdd(&accB[m][mrow],      acc_b0[r]);
        atomicAdd(&accB[m][mrow + 16], acc_b1[r]);
    }
    __syncthreads();

    // message = S*w_s + B*w_b
    for (int t = threadIdx.x; t < 16 * CC; t += 256) {
        int m = t / CC, c = t % CC;
        smsg[m][c] = accS[m][c] * sw[c] + accB[m][c] * bw[c];
    }
    __syncthreads();

    // pairwise = msg @ compat^T ; q = u - pairwise
    for (int t = threadIdx.x; t < 16 * CC; t += 256) {
        int m = t / CC, c = t % CC;
        int n = i0 + m;
        float p = 0.0f;
        #pragma unroll
        for (int cp = 0; cp < CC; ++cp) p += smsg[m][cp] * scompat[c * CC + cp];
        qout[n * CC + c] = u[n * CC + c] - p;
    }
}

// ---------------------------------------------------------------------------
extern "C" void kernel_launch(void* const* d_in, const int* in_sizes, int n_in,
                              void* d_out, int out_size, void* d_ws, size_t ws_size,
                              hipStream_t stream) {
    (void)in_sizes; (void)n_in; (void)out_size; (void)ws_size;

    const float* unaries = (const float*)d_in[0];
    const float* img     = (const float*)d_in[1];
    const float* sw      = (const float*)d_in[2];
    const float* bw      = (const float*)d_in[3];
    const float* compat  = (const float*)d_in[4];
    float*       out     = (float*)d_out;

    // Workspace carve-up (all offsets 256B aligned)
    char* ws = (char*)d_ws;
    size_t off = 0;
    auto take = [&](size_t bytes) { char* p = ws + off; off = (off + bytes + 255) & ~(size_t)255; return p; };
    unsigned short* Ks   = (unsigned short*)take((size_t)NPIX * NPIX * 2);  // 170 MB
    unsigned short* Kb   = (unsigned short*)take((size_t)NPIX * NPIX * 2);  // 170 MB
    unsigned short* qTb  = (unsigned short*)take((size_t)CPAD * NPIX * 2);  // 576 KB
    float*          qbuf = (float*)take((size_t)NPIX * CC * 4);             // 756 KB
    float*          feat = (float*)take((size_t)9 * NPIX * 4);              // 324 KB

    feats_kernel<<<NPIX / 256, 256, 0, stream>>>(img, feat);
    buildk_kernel<<<dim3(NPIX / 512, NPIX), 256, 0, stream>>>(feat, (unsigned*)Ks, (unsigned*)Kb);

    for (int it = 0; it < ITERS; ++it) {
        const float* qsrc = (it == 0) ? unaries : qbuf;
        float*       qdst = (it == ITERS - 1) ? out : qbuf;
        softmax_kernel<<<NPIX / 256, 256, 0, stream>>>(qsrc, qTb);
        crf_step_kernel<<<NPIX / 16, 256, 0, stream>>>(Ks, Kb, qTb, unaries,
                                                       sw, bw, compat, qdst);
    }
}
